// GAT_27006754357768
// MI455X (gfx1250) — matrix operand order repaired and checked by reading, not compiled, not run on hardware
//
#include <hip/hip_runtime.h>
#include <hip/hip_bf16.h>
#include <math.h>

typedef __attribute__((ext_vector_type(16))) _Float16 v16h;
typedef __attribute__((ext_vector_type(8)))  _Float16 v8h;
typedef __attribute__((ext_vector_type(4)))  _Float16 v4h;
typedef __attribute__((ext_vector_type(8)))  float    v8f;

#define ACT_NONE 0
#define ACT_RELU 1
#define ACT_ELU  2

#define GB   32      // batch
#define GN   68      // nodes / sequence length
#define GNP  96      // GN padded up to a multiple of 32 (zero-filled K pad)
#define GF   512     // feature dim
#define GH   8       // GAT heads
#define GHID 512     // GAT hidden
#define DEC_H 4      // decoder heads
#define DH   128     // decoder head dim
#define FFD  2048    // FFN dim

// ---------------------------------------------------------------------------
// Staging. LDS layout:
//   sa [16][40]  halves : A tile [m][k], 80B pitch (16B-aligned fragment runs)
//   sbt[256][40] halves : B tile TRANSPOSED [n][k] -> fragment reads are
//                         contiguous 32B runs (ds_load_b128), conflict-free
// K is ALWAYS a multiple of 32 (host pads K=68 operands to 96 with zeros), so
// no K bounds. M/N edges use clamped UNCONDITIONAL loads (no selects): A rows
// >= M and B cols >= N only influence C entries that are never stored.
// ---------------------------------------------------------------------------
__device__ __forceinline__ void stage_a(
    _Float16 (*__restrict__ sa)[40], const float* __restrict__ Ab,
    int tM, int kb, int M, int lda, int tid, bool fast)
{
  if (fast) {
    int r  = tid >> 3;
    int c4 = (tid & 7) << 2;
    const float4 v = *(const float4*)(Ab + (long long)(tM + r) * lda + (kb + c4));
    v4h h; h[0] = (_Float16)v.x; h[1] = (_Float16)v.y;
           h[2] = (_Float16)v.z; h[3] = (_Float16)v.w;
    *(v4h*)&sa[r][c4] = h;                              // ds_store_b64
  } else {
    int c = tid & 31;
#pragma unroll
    for (int i = 0; i < 4; ++i) {
      int r  = (tid >> 5) + (i << 2);
      int gr = tM + r; gr = gr < M ? gr : (M - 1);      // clamp, rows>=M dont-care
      sa[r][c] = (_Float16)Ab[(long long)gr * lda + (kb + c)];
    }
  }
}

template<int TRANSB>
__device__ __forceinline__ void stage_b(
    _Float16 (*__restrict__ sbt)[40], const float* __restrict__ Bb,
    int tN, int kb, int Nn, int ldb, int tid, bool fast)
{
  if (fast) {
    if (TRANSB) {
      // B is N x K row-major: float4 along K -> contiguous ds_store_b64
#pragma unroll
      for (int i = 0; i < 16; ++i) {
        int q  = tid + (i << 7);
        int n  = q >> 3;
        int k4 = (q & 7) << 2;
        const float4 v = *(const float4*)(Bb + (long long)(tN + n) * ldb + (kb + k4));
        v4h h; h[0] = (_Float16)v.x; h[1] = (_Float16)v.y;
               h[2] = (_Float16)v.z; h[3] = (_Float16)v.w;
        *(v4h*)&sbt[n][k4] = h;
      }
    } else {
      // B is K x N row-major: float4 along N, scatter 4 rows of sbt
#pragma unroll
      for (int i = 0; i < 16; ++i) {
        int q  = tid + (i << 7);
        int r  = q >> 6;
        int c4 = (q & 63) << 2;
        const float4 v = *(const float4*)(Bb + (long long)(kb + r) * ldb + (tN + c4));
        sbt[c4 + 0][r] = (_Float16)v.x;
        sbt[c4 + 1][r] = (_Float16)v.y;
        sbt[c4 + 2][r] = (_Float16)v.z;
        sbt[c4 + 3][r] = (_Float16)v.w;
      }
    }
  } else {
#pragma unroll
    for (int i = 0; i < 64; ++i) {
      int e = tid + (i << 7);
      int k = e >> 8, n = e & 255;
      int gn = tN + n; gn = gn < Nn ? gn : (Nn - 1);    // clamp, cols>=Nn dont-care
      float v = TRANSB ? Bb[(long long)gn * ldb + (kb + k)]
                       : Bb[(long long)(kb + k) * ldb + gn];
      sbt[n][k] = (_Float16)v;
    }
  }
}

// ---------------------------------------------------------------------------
// Batched WMMA GEMM:  C[b] = act( alpha * A[b] @ B[b] + bias )
//   batch offsets: off = (b/div)*s1 + (b%div)*s2   (per matrix)
// 4 wave32 per workgroup; block tile 16(M) x 256(N); each wave owns 64 columns
// with 4 accumulators -> 4 WMMAs + 1 A-fragment read per barrier. All 8 B
// fragment ds_loads issue BEFORE the WMMA group so one dscnt wait covers them.
// Double-buffered LDS, one barrier per 32-wide K step. K % 32 == 0 required.
// ---------------------------------------------------------------------------
template<int TRANSB>
__global__ __launch_bounds__(128) void wmma_gemm_kernel(
    const float* __restrict__ A, const float* __restrict__ B,
    const float* __restrict__ bias, float* __restrict__ C,
    int M, int K, int Nn, int lda, int ldb, int ldc,
    int divA, long long sA1, long long sA2,
    int divB, long long sB1, long long sB2,
    int divC, long long sC1, long long sC2,
    float alpha, int act)
{
  const int tid  = threadIdx.x;
  const int wave = tid >> 5;
  const int lane = tid & 31;
  const int bidx = blockIdx.z;
  const int tM   = blockIdx.y * 16;
  const int tN   = blockIdx.x * 256;

  const float* Ab = A + (long long)(bidx / divA) * sA1 + (long long)(bidx % divA) * sA2;
  const float* Bb = B + (long long)(bidx / divB) * sB1 + (long long)(bidx % divB) * sB2;
  float*       Cb = C + (long long)(bidx / divC) * sC1 + (long long)(bidx % divC) * sC2;

  __shared__ _Float16 sa [2][16][40];
  __shared__ _Float16 sbt[2][256][40];

  v8f acc0 = {}, acc1 = {}, acc2 = {}, acc3 = {};
  const int m16   = lane & 15;
  const int hiSel = lane >> 4;        // lane group 0: K-lo/rows, 1: K-hi/rows+8
  const int aK0   = hiSel * 16;       // A fragment K offset (contiguous, matches B)
  const int bK0   = hiSel * 16;       // B fragment K offset
  const int nc0   = wave * 64 + m16;  // first owned column within 256-wide tile

  const bool fastA = (tM + 16 <= M)  && ((lda & 3) == 0);
  const bool fastB = (tN + 256 <= Nn) && ((ldb & 3) == 0);
  const int  nk    = K >> 5;          // K is a multiple of 32 by construction

  stage_a(sa[0], Ab, tM, 0, M, lda, tid, fastA);
  stage_b<TRANSB>(sbt[0], Bb, tN, 0, Nn, ldb, tid, fastB);

  int buf = 0;
  for (int t = 0; t < nk; ++t) {
    __syncthreads();
    if (t + 1 < nk) {                 // prefetch next K tile into other buffer
      int kb = (t + 1) << 5;
      stage_a(sa[buf ^ 1], Ab, tM, kb, M, lda, tid, fastA);
      stage_b<TRANSB>(sbt[buf ^ 1], Bb, tN, kb, Nn, ldb, tid, fastB);
    }
    // issue ALL fragment loads first (2 A + 8 B ds_load_b128), then 4 WMMAs
    v8h alo = *(const v8h*)&sa[buf][m16][aK0];
    v8h ahi = *(const v8h*)&sa[buf][m16][aK0 + 8];
    v8h b0l = *(const v8h*)&sbt[buf][nc0 +  0][bK0];
    v8h b0h = *(const v8h*)&sbt[buf][nc0 +  0][bK0 + 8];
    v8h b1l = *(const v8h*)&sbt[buf][nc0 + 16][bK0];
    v8h b1h = *(const v8h*)&sbt[buf][nc0 + 16][bK0 + 8];
    v8h b2l = *(const v8h*)&sbt[buf][nc0 + 32][bK0];
    v8h b2h = *(const v8h*)&sbt[buf][nc0 + 32][bK0 + 8];
    v8h b3l = *(const v8h*)&sbt[buf][nc0 + 48][bK0];
    v8h b3h = *(const v8h*)&sbt[buf][nc0 + 48][bK0 + 8];
    v16h af  = __builtin_shufflevector(alo, ahi, 0,1,2,3,4,5,6,7,8,9,10,11,12,13,14,15);
    v16h bf0 = __builtin_shufflevector(b0l, b0h, 0,1,2,3,4,5,6,7,8,9,10,11,12,13,14,15);
    v16h bf1 = __builtin_shufflevector(b1l, b1h, 0,1,2,3,4,5,6,7,8,9,10,11,12,13,14,15);
    v16h bf2 = __builtin_shufflevector(b2l, b2h, 0,1,2,3,4,5,6,7,8,9,10,11,12,13,14,15);
    v16h bf3 = __builtin_shufflevector(b3l, b3h, 0,1,2,3,4,5,6,7,8,9,10,11,12,13,14,15);
    acc0 = __builtin_amdgcn_wmma_f32_16x16x32_f16(false, af, false, bf0, (short)0, acc0, false, false);
    acc1 = __builtin_amdgcn_wmma_f32_16x16x32_f16(false, af, false, bf1, (short)0, acc1, false, false);
    acc2 = __builtin_amdgcn_wmma_f32_16x16x32_f16(false, af, false, bf2, (short)0, acc2, false, false);
    acc3 = __builtin_amdgcn_wmma_f32_16x16x32_f16(false, af, false, bf3, (short)0, acc3, false, false);
    buf ^= 1;
  }

#pragma unroll
  for (int j = 0; j < 4; ++j) {
    const v8f& acc = j == 0 ? acc0 : (j == 1 ? acc1 : (j == 2 ? acc2 : acc3));
    int gn = tN + nc0 + j * 16;
    if (gn < Nn) {
      float bv = bias ? bias[gn] : 0.0f;
#pragma unroll
      for (int r = 0; r < 8; ++r) {
        int gm = tM + r + hiSel * 8;  // C layout: VGPR r -> row r + 8*laneGroup
        if (gm < M) {
          float v = acc[r] * alpha + bv;
          if (act == ACT_RELU)      v = v > 0.f ? v : 0.f;
          else if (act == ACT_ELU)  v = v > 0.f ? v : expm1f(v);
          Cb[(long long)gm * ldc + gn] = v;
        }
      }
    }
  }
}

// ---------------------------------------------------------------------------
// GAT attention logits: w1[hb][n] = <Wh[hb][n][:], a1>, w2 likewise with a2
// Wh rows live at  Wh + hb*bstride + n*D  (bstride covers the K padding)
// ---------------------------------------------------------------------------
__global__ __launch_bounds__(256) void gat_logits_kernel(
    const float* __restrict__ Wh, const float* __restrict__ avec,
    float* __restrict__ w1, float* __restrict__ w2, int Nn, int D, long long bstride)
{
  int n = blockIdx.x, hb = blockIdx.y, tid = threadIdx.x;
  const float* row = Wh + (long long)hb * bstride + (long long)n * D;
  const float* a1  = avec + (long long)hb * 2 * D;
  const float* a2  = a1 + D;
  float s1 = 0.f, s2 = 0.f;
  for (int o = tid; o < D; o += 256) { float v = row[o]; s1 += v * a1[o]; s2 += v * a2[o]; }
  __shared__ float r1[256], r2[256];
  r1[tid] = s1; r2[tid] = s2; __syncthreads();
  for (int s = 128; s > 0; s >>= 1) {
    if (tid < s) { r1[tid] += r1[tid + s]; r2[tid] += r2[tid + s]; }
    __syncthreads();
  }
  if (tid == 0) { w1[(long long)hb * Nn + n] = r1[0]; w2[(long long)hb * Nn + n] = r2[0]; }
}

// ---------------------------------------------------------------------------
// GAT masked softmax over axis i (reference softmax axis=2 / axis=1):
// att[hb][i][j] = softmax_i( adj[b][i][j]>0 ? leaky(w1[i]+w2[j]) : -9e15 )
// att rows padded to `pitch` (pad columns pre-zeroed by host memset)
// ---------------------------------------------------------------------------
__global__ __launch_bounds__(128) void gat_softmax_col_kernel(
    const float* __restrict__ w1, const float* __restrict__ w2,
    const float* __restrict__ adj, float* __restrict__ att,
    int Nn, int pitch, int modAdj)
{
  int j = blockIdx.x, hb = blockIdx.y, tid = threadIdx.x;
  int b = hb % modAdj;
  __shared__ float red[128];
  bool on = tid < Nn;
  float val = -3.0e38f;
  if (on) {
    float e = w1[(long long)hb * Nn + tid] + w2[(long long)hb * Nn + j];
    e = e > 0.f ? e : 0.2f * e;
    float m = adj[(long long)b * Nn * Nn + (long long)tid * Nn + j];
    val = (m > 0.f) ? e : -9.0e15f;
  }
  red[tid] = val; __syncthreads();
  for (int s = 64; s > 0; s >>= 1) { if (tid < s) red[tid] = fmaxf(red[tid], red[tid + s]); __syncthreads(); }
  float mx = red[0]; __syncthreads();
  float ex = on ? expf(val - mx) : 0.f;
  red[tid] = ex; __syncthreads();
  for (int s = 64; s > 0; s >>= 1) { if (tid < s) red[tid] += red[tid + s]; __syncthreads(); }
  float inv = 1.0f / red[0];
  if (on) att[(long long)hb * Nn * pitch + (long long)tid * pitch + j] = ex * inv;
}

// Row softmax over `cols` entries of rows with `pitch` stride, in place.
__global__ __launch_bounds__(128) void softmax_row_kernel(
    float* __restrict__ s, int cols, int pitch)
{
  long long row = blockIdx.x;
  int tid = threadIdx.x;
  float* p = s + row * pitch;
  __shared__ float red[128];
  bool on = tid < cols;
  float v = on ? p[tid] : -3.0e38f;
  red[tid] = v; __syncthreads();
  for (int st = 64; st > 0; st >>= 1) { if (tid < st) red[tid] = fmaxf(red[tid], red[tid + st]); __syncthreads(); }
  float mx = red[0]; __syncthreads();
  float ex = on ? expf(v - mx) : 0.f;
  red[tid] = ex; __syncthreads();
  for (int st = 64; st > 0; st >>= 1) { if (tid < st) red[tid] += red[tid + st]; __syncthreads(); }
  if (on) p[tid] = ex / red[0];
}

// hp (H,B,N,512) -> xcat (B,N,H*512)
__global__ __launch_bounds__(256) void permute_hp_kernel(
    const float* __restrict__ hp, float* __restrict__ xcat)
{
  long long e = (long long)blockIdx.x * 256 + threadIdx.x;
  const long long total = (long long)GH * GB * GN * GHID;
  if (e >= total) return;
  int o = (int)(e & (GHID - 1));
  long long t = e >> 9;
  int n = (int)(t % GN); t /= GN;
  int b = (int)(t % GB);
  int h = (int)(t / GB);
  xcat[((long long)b * GN + n) * (GH * GHID) + (long long)h * GHID + o] = hp[e];
}

// out[row] = LayerNorm(x[row] + r[row]) * g + be     (D = 512, block = 256)
__global__ __launch_bounds__(256) void add_ln_kernel(
    const float* __restrict__ x, const float* __restrict__ r,
    const float* __restrict__ g, const float* __restrict__ be,
    float* __restrict__ out)
{
  long long row = blockIdx.x;
  int tid = threadIdx.x;
  const float* xp = x + row * GF;
  const float* rp = r + row * GF;
  float v0 = xp[tid] + rp[tid];
  float v1 = xp[tid + 256] + rp[tid + 256];
  __shared__ float s1[256], s2[256];
  s1[tid] = v0 + v1; s2[tid] = v0 * v0 + v1 * v1;
  __syncthreads();
  for (int s = 128; s > 0; s >>= 1) {
    if (tid < s) { s1[tid] += s1[tid + s]; s2[tid] += s2[tid + s]; }
    __syncthreads();
  }
  float mean = s1[0] * (1.0f / GF);
  float var  = s2[0] * (1.0f / GF) - mean * mean;
  float rstd = rsqrtf(var + 1e-5f);
  out[row * GF + tid]       = (v0 - mean) * rstd * g[tid]       + be[tid];
  out[row * GF + tid + 256] = (v1 - mean) * rstd * g[tid + 256] + be[tid + 256];
}

// ---------------------------------------------------------------------------
// host-side helpers
// ---------------------------------------------------------------------------
static void launch_gemm(hipStream_t stream, int transB,
    const float* A, int divA, long long sA1, long long sA2, int lda,
    const float* B, int divB, long long sB1, long long sB2, int ldb,
    const float* bias,
    float* C, int divC, long long sC1, long long sC2, int ldc,
    int M, int K, int Nn, int batches, float alpha, int act)
{
  dim3 grid((unsigned)((Nn + 255) / 256), (unsigned)((M + 15) / 16), (unsigned)batches);
  if (transB)
    wmma_gemm_kernel<1><<<grid, 128, 0, stream>>>(A, B, bias, C, M, K, Nn, lda, ldb, ldc,
        divA, sA1, sA2, divB, sB1, sB2, divC, sC1, sC2, alpha, act);
  else
    wmma_gemm_kernel<0><<<grid, 128, 0, stream>>>(A, B, bias, C, M, K, Nn, lda, ldb, ldc,
        divA, sA1, sA2, divB, sB1, sB2, divC, sC1, sC2, alpha, act);
}

// torch-style MHA with q_in == kv_in; weights at d_in[pbase..pbase+7].
// t3 (v) and sc (scores) use K-padded layouts (pads pre-zeroed by host).
static void run_mha(hipStream_t stream, void* const* d_in, int pbase,
                    const float* yin, float* t1, float* t2, float* t3,
                    float* sc, float* aout, float* mo)
{
  const float* Wq = (const float*)d_in[pbase + 0]; const float* bq = (const float*)d_in[pbase + 1];
  const float* Wk = (const float*)d_in[pbase + 2]; const float* bk = (const float*)d_in[pbase + 3];
  const float* Wv = (const float*)d_in[pbase + 4]; const float* bv = (const float*)d_in[pbase + 5];
  const float* Wo = (const float*)d_in[pbase + 6]; const float* bo = (const float*)d_in[pbase + 7];
  const long long SROW  = (long long)GN * GF;     // unpadded (B, 68, 512)
  const long long SROWP = (long long)GNP * GF;    // padded   (B, 96, 512)

  launch_gemm(stream, 0, yin, 1, SROW, 0, GF, Wq, 1, 0, 0, GF, bq,
              t1, 1, SROW, 0, GF, GN, GF, GF, GB, 1.0f, ACT_NONE);
  launch_gemm(stream, 0, yin, 1, SROW, 0, GF, Wk, 1, 0, 0, GF, bk,
              t2, 1, SROW, 0, GF, GN, GF, GF, GB, 1.0f, ACT_NONE);
  launch_gemm(stream, 0, yin, 1, SROW, 0, GF, Wv, 1, 0, 0, GF, bv,
              t3, 1, SROWP, 0, GF, GN, GF, GF, GB, 1.0f, ACT_NONE);

  // scores = q @ k^T / sqrt(dh) — head slices via (div=4, s1=row-stride, s2=128)
  launch_gemm(stream, 1, t1, DEC_H, SROW, DH, GF,
              t2, DEC_H, SROW, DH, GF, nullptr,
              sc, 1, (long long)GN * GNP, 0, GNP,
              GN, DH, GN, GB * DEC_H, 1.0f / sqrtf((float)DH), ACT_NONE);
  softmax_row_kernel<<<GB * DEC_H * GN, 128, 0, stream>>>(sc, GN, GNP);

  // out = attn @ v   (K padded to 96; sc pad cols are zero)
  launch_gemm(stream, 0, sc, 1, (long long)GN * GNP, 0, GNP,
              t3, DEC_H, SROWP, DH, GF, nullptr,
              aout, DEC_H, SROW, DH, GF,
              GN, GNP, DH, GB * DEC_H, 1.0f, ACT_NONE);

  // output projection
  launch_gemm(stream, 0, aout, 1, SROW, 0, GF, Wo, 1, 0, 0, GF, bo,
              mo, 1, SROW, 0, GF, GN, GF, GF, GB, 1.0f, ACT_NONE);
}

extern "C" void kernel_launch(void* const* d_in, const int* in_sizes, int n_in,
                              void* d_out, int out_size, void* d_ws, size_t ws_size,
                              hipStream_t stream) {
  (void)in_sizes; (void)n_in; (void)out_size; (void)ws_size;
  const float* x       = (const float*)d_in[0];
  const float* adj     = (const float*)d_in[1];
  const float* W_heads = (const float*)d_in[2];
  const float* a_heads = (const float*)d_in[3];
  const float* W_out   = (const float*)d_in[4];
  const float* a_out   = (const float*)d_in[5];
  const float* fc1_w   = (const float*)d_in[58];
  const float* fc1_b   = (const float*)d_in[59];
  const float* fc2_w   = (const float*)d_in[60];
  const float* fc2_b   = (const float*)d_in[61];
  float* out = (float*)d_out;

  // workspace layout (~190 MB of fp32); K-padded shapes noted
  float* ws = (float*)d_ws;
  size_t off = 0;
  auto alloc = [&](size_t nf) { float* p = ws + off; off += nf; return p; };
  float* Wh   = alloc((size_t)GH * GB * GNP * GHID);   // (H*B, 96, 512)  padded rows
  float* hp   = alloc((size_t)GH * GB * GN * GHID);
  float* xcat = alloc((size_t)GB * GN * GH * GHID);
  float* w1   = alloc((size_t)GH * GB * GN);
  float* w2   = alloc((size_t)GH * GB * GN);
  float* attH = alloc((size_t)GH * GB * GN * GNP);     // (H*B, 68, 96)   padded cols
  float* Who  = alloc((size_t)GB * GNP * GHID);        // (B, 96, 512)    padded rows
  float* ao   = alloc((size_t)GB * GN * GNP);          // (B, 68, 96)     padded cols
  float* y    = alloc((size_t)GB * GN * GF);
  float* y2   = alloc((size_t)GB * GN * GF);
  float* t1   = alloc((size_t)GB * GN * GF);
  float* t2   = alloc((size_t)GB * GN * GF);
  float* t3   = alloc((size_t)GB * GNP * GF);          // (B, 96, 512)    padded rows
  float* sc   = alloc((size_t)GB * DEC_H * GN * GNP);  // (B*4, 68, 96)   padded cols
  float* aout = alloc((size_t)GB * GN * GF);
  float* mo   = alloc((size_t)GB * GN * GF);
  float* ff   = alloc((size_t)GB * GN * FFD);
  float* h1   = alloc((size_t)GB * GN * 64);

  // zero the K-padded operands once per launch (pads must contribute 0)
  hipMemsetAsync(Wh,   0, (size_t)GH * GB * GNP * GHID  * 4, stream);
  hipMemsetAsync(attH, 0, (size_t)GH * GB * GN * GNP    * 4, stream);
  hipMemsetAsync(Who,  0, (size_t)GB * GNP * GHID       * 4, stream);
  hipMemsetAsync(ao,   0, (size_t)GB * GN * GNP         * 4, stream);
  hipMemsetAsync(t3,   0, (size_t)GB * GNP * GF         * 4, stream);
  hipMemsetAsync(sc,   0, (size_t)GB * DEC_H * GN * GNP * 4, stream);

  // ---- GAT heads: Wh[hb] = x[b] @ W_heads[hb]   (hb = h*32+b -> A off = (hb%32)*stride)
  launch_gemm(stream, 0, x, GB, 0, (long long)GN * GF, GF,
              W_heads, 1, (long long)GF * GHID, 0, GHID, nullptr,
              Wh, 1, (long long)GNP * GHID, 0, GHID,
              GN, GF, GHID, GH * GB, 1.0f, ACT_NONE);
  gat_logits_kernel<<<dim3(GN, GH * GB), 256, 0, stream>>>(
      Wh, a_heads, w1, w2, GN, GHID, (long long)GNP * GHID);
  gat_softmax_col_kernel<<<dim3(GN, GH * GB), 128, 0, stream>>>(
      w1, w2, adj, attH, GN, GNP, GB);
  // hp = elu(att @ Wh)   (K padded to 96; attH pad cols & Wh pad rows are zero)
  launch_gemm(stream, 0, attH, 1, (long long)GN * GNP, 0, GNP,
              Wh, 1, (long long)GNP * GHID, 0, GHID, nullptr,
              hp, 1, (long long)GN * GHID, 0, GHID,
              GN, GNP, GHID, GH * GB, 1.0f, ACT_ELU);
  {
    long long total = (long long)GH * GB * GN * GHID;
    permute_hp_kernel<<<(unsigned)((total + 255) / 256), 256, 0, stream>>>(hp, xcat);
  }

  // ---- output GAT layer
  launch_gemm(stream, 0, xcat, 1, (long long)GN * GH * GHID, 0, GH * GHID,
              W_out, 1, (long long)GH * GHID * GHID, 0, GHID, nullptr,
              Who, 1, (long long)GNP * GHID, 0, GHID,
              GN, GH * GHID, GHID, GB, 1.0f, ACT_NONE);
  gat_logits_kernel<<<dim3(GN, GB), 256, 0, stream>>>(
      Who, a_out, w1, w2, GN, GHID, (long long)GNP * GHID);
  gat_softmax_col_kernel<<<dim3(GN, GB), 128, 0, stream>>>(
      w1, w2, adj, ao, GN, GNP, GB);
  launch_gemm(stream, 0, ao, 1, (long long)GN * GNP, 0, GNP,
              Who, 1, (long long)GNP * GHID, 0, GHID, nullptr,
              y, 1, (long long)GN * GHID, 0, GHID,
              GN, GNP, GHID, GB, 1.0f, ACT_NONE);

  // ---- transformer decoder, 2 layers (post-LN, relu FFN)
  for (int L = 0; L < 2; ++L) {
    int p = 6 + L * 26;
    run_mha(stream, d_in, p + 0, y, t1, t2, t3, sc, aout, mo);             // self-attn
    add_ln_kernel<<<GB * GN, 256, 0, stream>>>(mo, y,
        (const float*)d_in[p + 16], (const float*)d_in[p + 17], y2);       // ln1
    run_mha(stream, d_in, p + 8, y2, t1, t2, t3, sc, aout, mo);            // "cross"-attn
    add_ln_kernel<<<GB * GN, 256, 0, stream>>>(mo, y2,
        (const float*)d_in[p + 18], (const float*)d_in[p + 19], y);        // ln2
    // FFN
    launch_gemm(stream, 0, y, 1, (long long)GN * GF, 0, GF,
                (const float*)d_in[p + 22], 1, 0, 0, FFD, (const float*)d_in[p + 23],
                ff, 1, (long long)GN * FFD, 0, FFD,
                GN, GF, FFD, GB, 1.0f, ACT_RELU);
    launch_gemm(stream, 0, ff, 1, (long long)GN * FFD, 0, FFD,
                (const float*)d_in[p + 24], 1, 0, 0, GF, (const float*)d_in[p + 25],
                mo, 1, (long long)GN * GF, 0, GF,
                GN, FFD, GF, GB, 1.0f, ACT_NONE);
    add_ln_kernel<<<GB * GN, 256, 0, stream>>>(mo, y,
        (const float*)d_in[p + 20], (const float*)d_in[p + 21], y);        // ln3
  }

  // ---- MLP head
  launch_gemm(stream, 0, y, 1, (long long)GN * GF, 0, GF,
              fc1_w, 1, 0, 0, 64, fc1_b,
              h1, 1, (long long)GN * 64, 0, 64,
              GN, GF, 64, GB, 1.0f, ACT_RELU);
  launch_gemm(stream, 0, h1, 1, (long long)GN * 64, 0, 64,
              fc2_w, 1, 0, 0, 53, fc2_b,
              out, 1, (long long)GN * 53, 0, 53,
              GN, 64, 53, GB, 1.0f, ACT_NONE);
}
